// GCN_37563783970969
// MI455X (gfx1250) — matrix operand order repaired and checked by reading, not compiled, not run on hardware
//
#include <hip/hip_runtime.h>
#include <hip/hip_bf16.h>
#include <stdint.h>

// ---- CDNA5 vector types -----------------------------------------------------
typedef __attribute__((ext_vector_type(16))) __bf16        v16bf;
typedef __attribute__((ext_vector_type(8)))  float         v8f;
typedef __attribute__((ext_vector_type(4)))  unsigned int  u32x4;

struct H16 { u32x4 lo, hi; };

__device__ __forceinline__ v16bf mk16(u32x4 lo, u32x4 hi) {
  H16 h; h.lo = lo; h.hi = hi;
  return __builtin_bit_cast(v16bf, h);
}

__device__ __forceinline__ unsigned short f2bf(float f) {  // RNE fp32 -> bf16
  union { float f; unsigned u; } v; v.f = f;
  unsigned u = v.u;
  u = (u + 0x7FFFu + ((u >> 16) & 1u)) >> 16;
  return (unsigned short)u;
}
__device__ __forceinline__ float bf2f(unsigned short s) {
  union { unsigned u; float f; } v; v.u = ((unsigned)s) << 16;
  return v.f;
}

// Problem constants
#define NB   256     // batch
#define NE   63      // entities
#define NR   128     // relations
#define ND   512     // D
#define NP   1000    // params rows
#define NPP  1024    // params rows padded (zero rows 1000..1023)
#define NROW 254     // R + 2E
#define NX   65024   // B * 254

// ---- fp32 -> bf16 conversion (optionally fused relu) ------------------------
template<bool RELU>
__global__ void cvt_kernel(const float* __restrict__ s, unsigned short* __restrict__ d, int n) {
  int i = blockIdx.x * blockDim.x + threadIdx.x;
  int st = gridDim.x * blockDim.x;
  for (; i < n; i += st) {
    float v = s[i];
    if (RELU) v = v > 0.f ? v : 0.f;
    d[i] = f2bf(v);
  }
}

__global__ void zero_kernel(unsigned short* __restrict__ d, int n) {
  int i = blockIdx.x * blockDim.x + threadIdx.x;
  int st = gridDim.x * blockDim.x;
  for (; i < n; i += st) d[i] = 0;
}

// params (1000x512) -> params^T bf16 (512 x 1024, zero-padded cols 1000..1023)
__global__ void transpose_kernel(const float* __restrict__ P, unsigned short* __restrict__ PT) {
  int i = blockIdx.x * blockDim.x + threadIdx.x;
  int st = gridDim.x * blockDim.x;
  for (; i < ND * NPP; i += st) {
    int dd = i >> 10, pp = i & (NPP - 1);
    PT[i] = (pp < NP) ? f2bf(P[(size_t)pp * ND + dd]) : (unsigned short)0;
  }
}

// ---- fused gather + bf16 WMMA GEMM ------------------------------------------
// Block: 256 threads (8 wave32), tile 128(M) x 64(N); each wave owns a 16x64
// strip = four 16x16 WMMA accumulators sharing one A-fragment per K-chunk.
// B tiles (32x64) are double-buffered in LDS: next tile's global loads are
// issued before the wmma block, stored to the alternate buffer after it,
// one barrier per K-chunk.
// AMAP 0: A row m -> [W_word[i0] | W_word[i1]]                     (K = 1024)
// AMAP 1: A row m -> [W_word[j0] | W_word[j1] | relu(W_rela[j2])]  (K = 1536)
// EPI  0: relu(acc+b) -> x rows b*254+128+e   (+ bf16 copy)
// EPI  1: relu(acc+b) -> x_cat rows b*256+rowoff+r (fp32)
// EPI  2: 0.5*(W_word[j1] + relu(acc+b)) -> x rows b*254+r (+ bf16 copy)
template<int AMAP, int EPI>
__global__ void __launch_bounds__(256) gemm_fused(
    const int* __restrict__ idx,
    const unsigned short* __restrict__ Ww,    // W_word bf16
    const unsigned short* __restrict__ Wr,    // relu(W_rela) bf16
    const unsigned short* __restrict__ Bw,    // weight bf16, K x 512 row-major
    const float* __restrict__ bias,
    const float* __restrict__ WwF,            // W_word fp32 (EPI 2)
    float* __restrict__ outF,
    unsigned short* __restrict__ outB,
    float* __restrict__ xcat,
    int rowoff, int K)
{
  __shared__ __align__(16) unsigned short Bt[2][64 * 32];  // B tiles, [n][k]
  const int tid  = threadIdx.x;
  const int wave = tid >> 5;
  const int lane = tid & 31;
  const int mBase = blockIdx.x * 128 + wave * 16;
  const int nBase = blockIdx.y * 64;

  // Per-lane gathered A-row base pointers (fixed for the whole K loop).
  const int mA = mBase + (lane & 15);
  const unsigned short *p0, *p1, *p2 = nullptr;
  if (AMAP == 0) {
    int i0 = idx[(size_t)mA * 2 + 0];
    int i1 = idx[(size_t)mA * 2 + 1];
    p0 = Ww + (size_t)i0 * ND;
    p1 = Ww + (size_t)i1 * ND;
  } else {
    int j0 = idx[(size_t)mA * 3 + 0];
    int j1 = idx[(size_t)mA * 3 + 1];
    int j2 = idx[(size_t)mA * 3 + 2];
    p0 = Ww + (size_t)j0 * ND;
    p1 = Ww + (size_t)j1 * ND;
    p2 = Wr + (size_t)j2 * ND;
  }
  const int kA = (lane >> 4) * 8;    // A frag K sub-offset (ISA 16-bit A layout)
  const int kB = (lane >> 4) * 8;    // B frag K sub-offset
  const int nL = lane & 31 & 15;

  const int h0 = tid * 2;            // even half index within 32x64 tile (k*64+n)
  unsigned int st[4];

  auto loadB = [&](int k0) {
#pragma unroll
    for (int g = 0; g < 4; ++g) {
      int h = h0 + g * 512;
      int kk = h >> 6, n0 = h & 63;
      st[g] = *reinterpret_cast<const unsigned int*>(
          Bw + (size_t)(k0 + kk) * ND + nBase + n0);
    }
  };
  auto storeB = [&](int buf) {
#pragma unroll
    for (int g = 0; g < 4; ++g) {
      int h = h0 + g * 512;
      int kk = h >> 6, n0 = h & 63;
      Bt[buf][n0 * 32 + kk]       = (unsigned short)(st[g] & 0xFFFFu);
      Bt[buf][(n0 + 1) * 32 + kk] = (unsigned short)(st[g] >> 16);
    }
  };

  v8f acc[4] = {};
  loadB(0);
  storeB(0);
  __syncthreads();

  const int nch = K >> 5;
  for (int c = 0; c < nch; ++c) {
    const int k0 = c << 5;
    const bool more = (c + 1 < nch);
    if (more) loadB(k0 + 32);          // next tile's loads in flight during math

    int seg = k0 >> 9;
    const unsigned short* base = (AMAP == 0)
        ? (seg ? p1 : p0)
        : (seg == 0 ? p0 : (seg == 1 ? p1 : p2));
    int kl = (k0 & 511) + kA;
    v16bf a = mk16(*reinterpret_cast<const u32x4*>(base + kl),
                   *reinterpret_cast<const u32x4*>(base + kl + 16));
    const unsigned short* bb = &Bt[c & 1][0];
#pragma unroll
    for (int q = 0; q < 4; ++q) {
      const unsigned short* lp = bb + (nL + 16 * q) * 32 + kB;
      v16bf b = mk16(*reinterpret_cast<const u32x4*>(lp),
                     *reinterpret_cast<const u32x4*>(lp + 16));
      acc[q] = __builtin_amdgcn_wmma_f32_16x16x32_bf16(false, a, false, b,
                                                       (short)0, acc[q], false, false);
    }
    if (more) storeB((c + 1) & 1);
    __syncthreads();
  }

#pragma unroll
  for (int q = 0; q < 4; ++q) {
    const int n = nBase + nL + q * 16;
    const float bn = bias[n];
#pragma unroll
    for (int i = 0; i < 8; ++i) {
      int m = mBase + (lane >> 4) * 8 + i;  // C/D layout: lanes16-31 hold M=8..15
      float v = acc[q][i] + bn;
      v = v > 0.f ? v : 0.f;
      if (EPI == 0) {
        int bb2 = m / NE, e = m % NE;
        size_t o = ((size_t)bb2 * NROW + NR + e) * ND + n;
        outF[o] = v; outB[o] = f2bf(v);
      } else if (EPI == 1) {
        int bb2 = m >> 7, r = m & 127;
        xcat[((size_t)bb2 * 256 + rowoff + r) * ND + n] = v;
      } else {
        int bb2 = m >> 7, r = m & 127;
        int j1 = idx[(size_t)m * 3 + 1];
        float vv = 0.5f * (WwF[(size_t)j1 * ND + n] + v);
        size_t o = ((size_t)bb2 * NROW + r) * ND + n;
        outF[o] = vv; outB[o] = f2bf(vv);
      }
    }
  }
}

// ---- x_enti: sparse neighbor average ---------------------------------------
__global__ void __launch_bounds__(256) enti_kernel(
    const int* __restrict__ enti, const int* __restrict__ sub,
    const float* __restrict__ xcat, const float* __restrict__ WwF,
    float* __restrict__ outF, unsigned short* __restrict__ outB)
{
  __shared__ int list[256];
  __shared__ int cnt;
  const int bb = blockIdx.x / NE;
  const int e  = blockIdx.x % NE;
  const int tid = threadIdx.x;
  const int i0 = enti[((size_t)bb * NE + e) * 2];
  if (tid == 0) cnt = 0;
  __syncthreads();
  {
    int p = tid;  // 0..255 over concat(sub[:,0], sub[:,1])
    int pv = (p < NR) ? sub[((size_t)bb * NR + p) * 3 + 0]
                      : sub[((size_t)bb * NR + (p - NR)) * 3 + 1];
    if (pv == i0) { int s = atomicAdd(&cnt, 1); list[s] = p; }
  }
  __syncthreads();
  const int c = cnt;
  const float inv = 1.0f / (float)(c + 1);
  for (int d = tid; d < ND; d += 256) {
    float v = WwF[(size_t)i0 * ND + d];              // obj_vector
    for (int t = 0; t < c; ++t)
      v += xcat[((size_t)bb * 256 + list[t]) * ND + d];
    v *= inv;
    size_t o = ((size_t)bb * NROW + NR + NE + e) * ND + d;
    outF[o] = v; outB[o] = f2bf(v);
  }
}

// ---- fused softmax attention: x -> relu(softmax(x P^T) P) + x, masked -------
// params is zero-padded to 1024 rows and params^T to 1024 cols, so every
// fragment load in both stages is an unconditional pair of b128 loads.
__global__ void __launch_bounds__(128) attn_kernel(
    const unsigned short* __restrict__ xB,    // x bf16, NX x 512
    const unsigned short* __restrict__ Pbf,   // params bf16, 1024 x 512 (padded)
    const unsigned short* __restrict__ PTbf,  // params^T bf16, 512 x 1024 (padded)
    float* __restrict__ xF,                   // d_out x (read + overwrite)
    const unsigned char* __restrict__ sg0,
    const unsigned char* __restrict__ sg1)
{
  __shared__ __align__(16) unsigned short xT[16 * 528];   // x tile bf16
  __shared__ __align__(16) unsigned short S[16 * 1040];   // logits -> probs bf16
  __shared__ float red[16 * 8];
  __shared__ float rowMax[16], rowSum[16];

  const int tid  = threadIdx.x;
  const int wave = tid >> 5;
  const int lane = tid & 31;
  const int rowBase = blockIdx.x * 16;
  const int kA = (lane >> 4) * 8;
  const int kB = (lane >> 4) * 8;

  // load 16x512 x-tile (b128 loads/stores)
  for (int v = tid; v < 1024; v += 128) {
    int e0 = v * 8;
    int m = e0 >> 9, k = e0 & 511;
    u32x4 d = *reinterpret_cast<const u32x4*>(xB + (size_t)(rowBase + m) * ND + k);
    *reinterpret_cast<u32x4*>(&xT[m * 528 + k]) = d;
  }
  // zero K padding 1008..1039 (cols 1000..1007 are written as exact zeros by
  // stage 1 since params rows 1000..1023 are zero)
  for (int v = tid; v < 16 * 32; v += 128) {
    int m = v >> 5, c = v & 31;
    S[m * 1040 + 1008 + c] = 0;
  }
  __syncthreads();

  // stage 1: S = x_tile @ params^T  (16 x 1008, cols >=1000 exactly zero)
  for (int nt = wave; nt < 63; nt += 4) {
    int n = nt * 16 + (lane & 15);          // <= 1007 < padded 1024 rows
    const unsigned short* brow = Pbf + (size_t)n * ND;
    v8f acc = {};
    int ma = lane & 15;
    for (int k0 = 0; k0 < ND; k0 += 32) {
      const unsigned short* ap = &xT[ma * 528 + k0 + kA];
      v16bf a = mk16(*reinterpret_cast<const u32x4*>(ap),
                     *reinterpret_cast<const u32x4*>(ap + 16));
      const unsigned short* bp = brow + k0 + kB;
      v16bf b = mk16(*reinterpret_cast<const u32x4*>(bp),
                     *reinterpret_cast<const u32x4*>(bp + 16));
      acc = __builtin_amdgcn_wmma_f32_16x16x32_bf16(false, a, false, b,
                                                    (short)0, acc, false, false);
    }
#pragma unroll
    for (int i = 0; i < 8; ++i) {
      int m = (lane >> 4) * 8 + i;
      S[m * 1040 + n] = f2bf(acc[i]);
    }
  }
  __syncthreads();

  // softmax over 1000 cols per row (8 threads per row)
  {
    int row = tid >> 3, j = tid & 7;
    float lmax = -3.4e38f;
    for (int c = j; c < NP; c += 8)
      lmax = fmaxf(lmax, bf2f(S[row * 1040 + c]));
    red[row * 8 + j] = lmax;
    __syncthreads();
    if (j == 0) {
      float m = red[row * 8];
      for (int t = 1; t < 8; ++t) m = fmaxf(m, red[row * 8 + t]);
      rowMax[row] = m;
    }
    __syncthreads();
    float mx = rowMax[row];
    float lsum = 0.f;
    for (int c = j; c < NP; c += 8) {
      float e = __expf(bf2f(S[row * 1040 + c]) - mx);
      S[row * 1040 + c] = f2bf(e);
      lsum += e;
    }
    red[row * 8 + j] = lsum;
    __syncthreads();
    if (j == 0) {
      float s = 0.f;
      for (int t = 0; t < 8; ++t) s += red[row * 8 + t];
      rowSum[row] = s;
    }
    __syncthreads();
    float inv = 1.0f / rowSum[row];
    for (int c = j; c < NP; c += 8)
      S[row * 1040 + c] = f2bf(bf2f(S[row * 1040 + c]) * inv);
  }
  __syncthreads();

  // stage 2: out = relu(probs @ params) + x, masked  (K padded to 1024)
  for (int ct = wave; ct < 32; ct += 4) {
    int d = ct * 16 + (lane & 15);
    const unsigned short* bcol = PTbf + (size_t)d * NPP;
    v8f acc = {};
    int ma = lane & 15;
    for (int k0 = 0; k0 < NPP; k0 += 32) {
      const unsigned short* ap = &S[ma * 1040 + k0 + kA];
      v16bf a = mk16(*reinterpret_cast<const u32x4*>(ap),
                     *reinterpret_cast<const u32x4*>(ap + 16));
      const unsigned short* bp = bcol + k0 + kB;
      v16bf b = mk16(*reinterpret_cast<const u32x4*>(bp),
                     *reinterpret_cast<const u32x4*>(bp + 16));
      acc = __builtin_amdgcn_wmma_f32_16x16x32_bf16(false, a, false, b,
                                                    (short)0, acc, false, false);
    }
#pragma unroll
    for (int i = 0; i < 8; ++i) {
      int m = (lane >> 4) * 8 + i;
      int row = rowBase + m;
      int bb = row / NROW, rr = row % NROW;
      unsigned char mk;
      if (rr < NR)            mk = sg0[(size_t)bb * NR + rr];
      else if (rr < NR + NE)  mk = sg1[(size_t)bb * NE + (rr - NR)];
      else                    mk = sg1[(size_t)bb * NE + (rr - NR - NE)];
      size_t o = (size_t)row * ND + d;
      float v = fmaxf(acc[i], 0.f) + xF[o];
      xF[o] = mk ? 0.f : v;
    }
  }
}

// ---- mask output ------------------------------------------------------------
__global__ void mask_kernel(const unsigned char* __restrict__ sg0,
                            const unsigned char* __restrict__ sg1,
                            float* __restrict__ outM) {
  int i = blockIdx.x * blockDim.x + threadIdx.x;
  if (i >= NX) return;
  int bb = i / NROW, rr = i % NROW;
  unsigned char mk;
  if (rr < NR)            mk = sg0[bb * NR + rr];
  else if (rr < NR + NE)  mk = sg1[bb * NE + (rr - NR)];
  else                    mk = sg1[bb * NE + (rr - NR - NE)];
  outM[i] = mk ? 1.0f : 0.0f;
}

// ---- launch -----------------------------------------------------------------
extern "C" void kernel_launch(void* const* d_in, const int* in_sizes, int n_in,
                              void* d_out, int out_size, void* d_ws, size_t ws_size,
                              hipStream_t stream) {
  (void)in_sizes; (void)n_in; (void)out_size; (void)ws_size;
  const int*   enti  = (const int*)d_in[0];
  const int*   sub   = (const int*)d_in[1];
  const unsigned char* sg0 = (const unsigned char*)d_in[2];
  const unsigned char* sg1 = (const unsigned char*)d_in[3];
  const float* Wword = (const float*)d_in[4];
  const float* Wrel  = (const float*)d_in[5];
  const float* params= (const float*)d_in[6];
  const float* wattr = (const float*)d_in[7];
  const float* battr = (const float*)d_in[8];
  const float* wrela = (const float*)d_in[9];
  const float* brela = (const float*)d_in[10];
  const float* wsub  = (const float*)d_in[11];
  const float* bsub  = (const float*)d_in[12];
  const float* wobj  = (const float*)d_in[13];
  const float* bobj  = (const float*)d_in[14];

  char* ws = (char*)d_ws;
  size_t off = 0;
  auto alloc = [&](size_t bytes) -> char* {
    char* p = ws + off;
    off = (off + bytes + 255) & ~(size_t)255;
    return p;
  };
  unsigned short* WwB   = (unsigned short*)alloc((size_t)10000 * ND * 2);
  unsigned short* WrB   = (unsigned short*)alloc((size_t)472 * ND * 2);
  unsigned short* wattrB= (unsigned short*)alloc((size_t)1024 * ND * 2);
  unsigned short* wrelaB= (unsigned short*)alloc((size_t)1536 * ND * 2);
  unsigned short* wsubB = (unsigned short*)alloc((size_t)1536 * ND * 2);
  unsigned short* wobjB = (unsigned short*)alloc((size_t)1536 * ND * 2);
  unsigned short* PB    = (unsigned short*)alloc((size_t)NPP * ND * 2);  // padded
  unsigned short* PTB   = (unsigned short*)alloc((size_t)ND * NPP * 2);  // padded
  unsigned short* xBf   = (unsigned short*)alloc((size_t)NX * ND * 2);
  float*          xcat  = (float*)alloc((size_t)NB * 256 * ND * 4);

  float* xOut = (float*)d_out;                 // NX x 512 fp32
  float* mOut = xOut + (size_t)NX * ND;        // NX mask floats

  auto cblk = [](int n) { int g = (n + 255) / 256; return g > 16384 ? 16384 : g; };

  // weight conversions (relu folded into W_rela copy, commutes with gather)
  cvt_kernel<false><<<cblk(10000 * ND), 256, 0, stream>>>(Wword, WwB, 10000 * ND);
  cvt_kernel<true ><<<cblk(472 * ND),   256, 0, stream>>>(Wrel,  WrB,  472 * ND);
  cvt_kernel<false><<<cblk(1024 * ND),  256, 0, stream>>>(wattr, wattrB, 1024 * ND);
  cvt_kernel<false><<<cblk(1536 * ND),  256, 0, stream>>>(wrela, wrelaB, 1536 * ND);
  cvt_kernel<false><<<cblk(1536 * ND),  256, 0, stream>>>(wsub,  wsubB,  1536 * ND);
  cvt_kernel<false><<<cblk(1536 * ND),  256, 0, stream>>>(wobj,  wobjB,  1536 * ND);
  cvt_kernel<false><<<cblk(NP * ND),    256, 0, stream>>>(params, PB, NP * ND);
  zero_kernel<<<cblk((NPP - NP) * ND), 256, 0, stream>>>(PB + (size_t)NP * ND, (NPP - NP) * ND);
  transpose_kernel<<<cblk(ND * NPP), 256, 0, stream>>>(params, PTB);

  // x_attr : (B*E) x 1024 x 512
  gemm_fused<0, 0><<<dim3(126, 8), 256, 0, stream>>>(
      enti, WwB, WrB, wattrB, battr, nullptr, xOut, xBf, nullptr, 0, 1024);
  // x_sub / x_obj : (B*R) x 1536 x 512 -> x_cat
  gemm_fused<1, 1><<<dim3(256, 8), 256, 0, stream>>>(
      sub, WwB, WrB, wsubB, bsub, nullptr, nullptr, nullptr, xcat, 0, 1536);
  gemm_fused<1, 1><<<dim3(256, 8), 256, 0, stream>>>(
      sub, WwB, WrB, wobjB, bobj, nullptr, nullptr, nullptr, xcat, 128, 1536);
  // x_rela : (B*R) x 1536 x 512 with residual-mean epilogue
  gemm_fused<1, 2><<<dim3(256, 8), 256, 0, stream>>>(
      sub, WwB, WrB, wrelaB, brela, Wword, xOut, xBf, nullptr, 0, 1536);

  // x_enti : sparse neighbor average
  enti_kernel<<<NB * NE, 256, 0, stream>>>(enti, sub, xcat, Wword, xOut, xBf);

  // fused softmax attention + residual + mask
  attn_kernel<<<NX / 16, 128, 0, stream>>>(xBf, PB, PTB, xOut, sg0, sg1);

  // mask output tail
  mask_kernel<<<(NX + 255) / 256, 256, 0, stream>>>(sg0, sg1, mOut);
}
